// Multi_Headed_59304908423973
// MI455X (gfx1250) — matrix operand order repaired and checked by reading, not compiled, not run on hardware
//
#include <hip/hip_runtime.h>
#include <hip/hip_bf16.h>
#include <stdint.h>

// Multi-head causal attention for MI455X (gfx1250, wave32, WMMA).
// Pipeline: [qkv_proj bf16] -> [flash attention bf16/f32, async-LDS double
// buffered K/V] -> [out proj f32].
// All matmuls use v_wmma_f32_16x16x32_bf16 with f32 accumulation.

#define EMB   768
#define HEADS 3
#define HD    256
#define BATCH 4
#define SEQ   2048
#define FA_WAVES 8                      // waves per flash-attention block

typedef __bf16 bf16_t;
typedef __attribute__((ext_vector_type(16))) __bf16 v16bf;
typedef __attribute__((ext_vector_type(8)))  float  v8f;

static __device__ __forceinline__ bf16_t f2bf(float f) { return (bf16_t)f; }

union PairU { unsigned int u; bf16_t h[2]; };

// A-fragment (16x32 bf16) K-index for packed pair j (0..7), lane group g:
//  lanes 0-15 (g=0): v0..v7 hold K=0..15
//  lanes16-31 (g=1): v0..v7 hold K=16..31
static __device__ __forceinline__ int a_k(int j, int g) {
  return g * 16 + 2 * j;
}

// Reductions across a 16-lane half-wave (rows of a D fragment live in one half)
static __device__ __forceinline__ float grp_max(float v) {
  v = fmaxf(v, __shfl_xor(v, 1, 32));
  v = fmaxf(v, __shfl_xor(v, 2, 32));
  v = fmaxf(v, __shfl_xor(v, 4, 32));
  v = fmaxf(v, __shfl_xor(v, 8, 32));
  return v;
}
static __device__ __forceinline__ float grp_sum(float v) {
  v += __shfl_xor(v, 1, 32);
  v += __shfl_xor(v, 2, 32);
  v += __shfl_xor(v, 4, 32);
  v += __shfl_xor(v, 8, 32);
  return v;
}

// ---------------------------------------------------------------------------
// Kernel 1: per-head QKV projection, X[b] (S x EMB, f32) * W[h] (EMB x HD, f32)
// -> bf16 [b*H+h][S][HD]. One wave per 16x16 output tile; f32->bf16 fused
// into the fragment load.
// ---------------------------------------------------------------------------
__global__ __launch_bounds__(32) void qkv_proj_kernel(
    const float* __restrict__ Xk, const float* __restrict__ Xv,
    const float* __restrict__ Xq,
    const float* __restrict__ Wk, const float* __restrict__ Wv,
    const float* __restrict__ Wq,
    bf16_t* __restrict__ Kb, bf16_t* __restrict__ Vb, bf16_t* __restrict__ Qb) {
  const int lane = threadIdx.x;
  const int n = lane & 15, g = lane >> 4;
  const int st = blockIdx.x;                 // seq tile (S/16)
  const int ft = blockIdx.y;                 // feature tile (HD/16)
  const int z  = blockIdx.z;                 // t*(B*H) + b*H + h
  const int t  = z / (BATCH * HEADS);
  const int bh = z % (BATCH * HEADS);
  const int h  = bh % HEADS;
  const int b  = bh / HEADS;

  const float* X = (t == 0 ? Xk : (t == 1 ? Xv : Xq)) + (size_t)b * SEQ * EMB;
  const float* W = (t == 0 ? Wk : (t == 1 ? Wv : Wq)) + (size_t)h * EMB * HD;
  bf16_t*      O = (t == 0 ? Kb : (t == 1 ? Vb : Qb)) + (size_t)bh * SEQ * HD;

  const int row = st * 16 + n;               // A-fragment row (lane&15)
  const int col = ft * 16 + n;               // B-fragment column
  const int bkb = g * 16;                    // B-fragment K base per half-wave

  v8f acc = {0.f, 0.f, 0.f, 0.f, 0.f, 0.f, 0.f, 0.f};
  for (int c = 0; c < EMB / 32; ++c) {
    const int kb = c * 32;
    v16bf a, w;
#pragma unroll
    for (int j = 0; j < 8; ++j) {
      const int ka = kb + a_k(j, g);
      const float2 xf = *reinterpret_cast<const float2*>(&X[(size_t)row * EMB + ka]);
      a[2 * j]     = f2bf(xf.x);
      a[2 * j + 1] = f2bf(xf.y);
      const int kw = kb + bkb + 2 * j;
      w[2 * j]     = f2bf(W[(size_t)kw * HD + col]);
      w[2 * j + 1] = f2bf(W[(size_t)(kw + 1) * HD + col]);
    }
    acc = __builtin_amdgcn_wmma_f32_16x16x32_bf16(false, a, false, w,
                                                  (short)0, acc, false, false);
  }
#pragma unroll
  for (int r = 0; r < 8; ++r) {
    const int dm = st * 16 + r + g * 8;      // D rows: lanes0-15 -> 0..7, lanes16-31 -> 8..15
    O[(size_t)dm * HD + ft * 16 + n] = f2bf(acc[r]);
  }
}

// ---------------------------------------------------------------------------
// Kernel 2: causal flash attention. One block = 8 waves = 128 query rows of
// one (b,h). K/V streamed in 32-key blocks, staged to LDS with
// global_load_async_to_lds_b128 (ASYNCcnt) and double buffered so the async
// copy of block i+1 overlaps the 32 WMMAs of block i.
// Per block: QK^T (16 WMMA) -> online softmax (f32, half-wave shfl row
// reductions) -> P transposed via per-wave LDS scratch -> PV (16 WMMA).
// Reference scale is 1/sqrt(S_keys) = 1/sqrt(2048).
// ---------------------------------------------------------------------------
__global__ __launch_bounds__(32 * FA_WAVES) void flash_attn_kernel(
    const bf16_t* __restrict__ Qb, const bf16_t* __restrict__ Kb,
    const bf16_t* __restrict__ Vb, bf16_t* __restrict__ Cc) {
  __shared__ __align__(16) bf16_t Kt[2][32 * HD];          // 2 x 16KB
  __shared__ __align__(16) bf16_t Vt[2][32 * HD];          // 2 x 16KB
  __shared__ __align__(16) bf16_t Pt[FA_WAVES][16 * 32];   // 8 x 1KB

  const int tid  = threadIdx.x;
  const int lane = tid & 31;
  const int wv   = tid >> 5;
  const int n = lane & 15, g = lane >> 4;
  const int qt = blockIdx.x * FA_WAVES + wv;  // this wave's query tile
  const int bh = blockIdx.y;                  // b*H + h
  const int h  = bh % HEADS;
  const int b  = bh / HEADS;

  const bf16_t* Qp = Qb + (size_t)bh * SEQ * HD;
  const bf16_t* Kp = Kb + (size_t)bh * SEQ * HD;
  const bf16_t* Vp = Vb + (size_t)bh * SEQ * HD;
  bf16_t*       Cp = Cc + (size_t)b * SEQ * EMB;

  const int q0  = qt * 16;
  const int bkb = g * 16;
  const float inv_scale = 0.022097086912079612f;   // 1/sqrt(2048)
  const v8f zero8 = {0.f, 0.f, 0.f, 0.f, 0.f, 0.f, 0.f, 0.f};

  // ---- preload this wave's Q tile (16 x 256) as 8 A fragments ----
  v16bf qf[8];
#pragma unroll
  for (int c = 0; c < HD / 32; ++c) {
#pragma unroll
    for (int j = 0; j < 8; ++j) {
      PairU pq;
      pq.u = *reinterpret_cast<const unsigned int*>(
          &Qp[(size_t)(q0 + n) * HD + c * 32 + a_k(j, g)]);
      qf[c][2 * j]     = pq.h[0];
      qf[c][2 * j + 1] = pq.h[1];
    }
  }

  float mrow[8], lrow[8];
  v8f Oacc[16];
#pragma unroll
  for (int r = 0; r < 8; ++r) { mrow[r] = -3.0e38f; lrow[r] = 0.f; }
#pragma unroll
  for (int ftI = 0; ftI < 16; ++ftI) Oacc[ftI] = zero8;

  // async stage of one 32-key K/V block (32 rows are contiguous 16KB panels)
  const unsigned int ldsK0 = (unsigned int)(uintptr_t)&Kt[0][0];
  const unsigned int ldsV0 = (unsigned int)(uintptr_t)&Vt[0][0];
  const unsigned int tileBytes = 32 * HD * 2;      // 16KB
  auto stage = [&](int buf, int k0s) {
    const unsigned long long gK = (unsigned long long)(uintptr_t)(Kp + (size_t)k0s * HD);
    const unsigned long long gV = (unsigned long long)(uintptr_t)(Vp + (size_t)k0s * HD);
    const unsigned int lK = ldsK0 + (unsigned int)buf * tileBytes;
    const unsigned int lV = ldsV0 + (unsigned int)buf * tileBytes;
#pragma unroll
    for (int i = 0; i < (int)(32 * HD * 2) / (16 * 32 * FA_WAVES); ++i) {  // 4
      const unsigned int off = (unsigned int)(tid * 16 + i * (32 * FA_WAVES * 16));
      asm volatile("global_load_async_to_lds_b128 %0, %1, %2"
                   :: "v"(lK + off), "v"(off), "s"(gK) : "memory");
      asm volatile("global_load_async_to_lds_b128 %0, %1, %2"
                   :: "v"(lV + off), "v"(off), "s"(gV) : "memory");
    }
  };

  // #32-key blocks covering the causal horizon of the whole block (8 q-tiles)
  const int nk32 = 4 * (blockIdx.x + 1);

  stage(0, 0);
  asm volatile("s_wait_asynccnt 0x0" ::: "memory");
  __syncthreads();

  for (int kb32 = 0; kb32 < nk32; ++kb32) {
    const int k0  = kb32 * 32;
    const int cur = kb32 & 1;

    // prefetch next block into the other buffer (its readers finished at the
    // barrier that ended the previous iteration)
    if (kb32 + 1 < nk32) stage(cur ^ 1, k0 + 32);

    if (k0 <= q0 + 15) {                     // wave-uniform causal horizon
      const bf16_t* Kc = &Kt[cur][0];
      const bf16_t* Vc = &Vt[cur][0];

      // ---- scores: two 16x16 D tiles (keys k0..+15 and k0+16..+31) ----
      v8f Sd[2];
#pragma unroll
      for (int stI = 0; stI < 2; ++stI) {
        v8f acc = zero8;
#pragma unroll
        for (int c = 0; c < HD / 32; ++c) {
          v16bf kf;
#pragma unroll
          for (int j = 0; j < 8; ++j) {
            PairU pk;   // B = K^T: column = key row; pairs contiguous in hd
            pk.u = *reinterpret_cast<const unsigned int*>(
                &Kc[(size_t)(stI * 16 + n) * HD + c * 32 + bkb + 2 * j]);
            kf[2 * j]     = pk.h[0];
            kf[2 * j + 1] = pk.h[1];
          }
          acc = __builtin_amdgcn_wmma_f32_16x16x32_bf16(false, qf[c], false, kf,
                                                        (short)0, acc, false, false);
        }
        Sd[stI] = acc;
      }

      // ---- online softmax update (f32) ----
      float p0a[8], p1a[8], scl[8];
#pragma unroll
      for (int r = 0; r < 8; ++r) {
        const int qg = q0 + r + g * 8;       // global query row of D element r
        float v0 = Sd[0][r] * inv_scale;
        float v1 = Sd[1][r] * inv_scale;
        if (k0 + n > qg)      v0 = -1.0e30f; // causal mask (k > q)
        if (k0 + 16 + n > qg) v1 = -1.0e30f;
        const float vm   = grp_max(fmaxf(v0, v1));
        const float mnew = fmaxf(mrow[r], vm);
        const float sc   = __expf(mrow[r] - mnew);
        const float p0   = __expf(v0 - mnew);
        const float p1   = __expf(v1 - mnew);
        const float rs   = grp_sum(p0 + p1);
        lrow[r] = lrow[r] * sc + rs;
        mrow[r] = mnew;
        scl[r]  = sc;
        p0a[r]  = p0;
        p1a[r]  = p1;
      }
#pragma unroll
      for (int ftI = 0; ftI < 16; ++ftI)
#pragma unroll
        for (int r = 0; r < 8; ++r) Oacc[ftI][r] *= scl[r];

      // ---- transpose P (D layout) -> A layout via per-wave LDS scratch
      // (in-wave DS ordering; no cross-wave sharing) ----
      bf16_t* Pw = &Pt[wv][0];
#pragma unroll
      for (int r = 0; r < 8; ++r) {
        Pw[(r + g * 8) * 32 + n]      = f2bf(p0a[r]);
        Pw[(r + g * 8) * 32 + 16 + n] = f2bf(p1a[r]);
      }
      v16bf pa;
#pragma unroll
      for (int j = 0; j < 8; ++j) {
        PairU pp;
        pp.u = *reinterpret_cast<const unsigned int*>(&Pw[n * 32 + a_k(j, g)]);
        pa[2 * j]     = pp.h[0];
        pa[2 * j + 1] = pp.h[1];
      }

      // ---- O += P x V over all 16 feature tiles of HD=256 ----
#pragma unroll
      for (int ftI = 0; ftI < 16; ++ftI) {
        v16bf vb;
        const int fcol = ftI * 16 + n;
#pragma unroll
        for (int j = 0; j < 8; ++j) {
          vb[2 * j]     = Vc[(size_t)(bkb + 2 * j)     * HD + fcol];
          vb[2 * j + 1] = Vc[(size_t)(bkb + 2 * j + 1) * HD + fcol];
        }
        Oacc[ftI] = __builtin_amdgcn_wmma_f32_16x16x32_bf16(false, pa, false, vb,
                                                            (short)0, Oacc[ftI],
                                                            false, false);
      }
    }

    // next buffer ready + all waves done reading current buffer
    asm volatile("s_wait_asynccnt 0x0" ::: "memory");
    __syncthreads();
  }

  // ---- epilogue: normalize and write concat-heads buffer (bf16) ----
#pragma unroll
  for (int ftI = 0; ftI < 16; ++ftI) {
#pragma unroll
    for (int r = 0; r < 8; ++r) {
      const float o = Oacc[ftI][r] / lrow[r];
      Cp[(size_t)(q0 + r + g * 8) * EMB + h * HD + ftI * 16 + n] = f2bf(o);
    }
  }
}

// ---------------------------------------------------------------------------
// Kernel 3: output projection  concat (S x EMB, bf16) * Wo (EMB x EMB, f32)
// + bo, f32 output.
// ---------------------------------------------------------------------------
__global__ __launch_bounds__(32) void out_proj_kernel(
    const bf16_t* __restrict__ Cc, const float* __restrict__ Wo,
    const float* __restrict__ bo, float* __restrict__ Out) {
  const int lane = threadIdx.x;
  const int n = lane & 15, g = lane >> 4;
  const int st = blockIdx.x;                 // seq tile (S/16)
  const int et = blockIdx.y;                 // emb tile (EMB/16)
  const int b  = blockIdx.z;

  const bf16_t* Cp = Cc + (size_t)b * SEQ * EMB;
  float*        Op = Out + (size_t)b * SEQ * EMB;

  const int row = st * 16 + n;
  const int col = et * 16 + n;
  const int bkb = g * 16;

  v8f acc = {0.f, 0.f, 0.f, 0.f, 0.f, 0.f, 0.f, 0.f};
  for (int c = 0; c < EMB / 32; ++c) {
    const int kb = c * 32;
    v16bf a, w;
#pragma unroll
    for (int j = 0; j < 8; ++j) {
      PairU pc;
      pc.u = *reinterpret_cast<const unsigned int*>(
          &Cp[(size_t)row * EMB + kb + a_k(j, g)]);
      a[2 * j]     = pc.h[0];
      a[2 * j + 1] = pc.h[1];
      const int kw = kb + bkb + 2 * j;
      w[2 * j]     = f2bf(Wo[(size_t)kw * EMB + col]);
      w[2 * j + 1] = f2bf(Wo[(size_t)(kw + 1) * EMB + col]);
    }
    acc = __builtin_amdgcn_wmma_f32_16x16x32_bf16(false, a, false, w,
                                                  (short)0, acc, false, false);
  }
  const float bias = bo[col];
#pragma unroll
  for (int r = 0; r < 8; ++r) {
    Op[(size_t)(st * 16 + r + g * 8) * EMB + col] = acc[r] + bias;
  }
}

// ---------------------------------------------------------------------------
extern "C" void kernel_launch(void* const* d_in, const int* in_sizes, int n_in,
                              void* d_out, int out_size, void* d_ws, size_t ws_size,
                              hipStream_t stream) {
  const float* Xk = (const float*)d_in[0];
  const float* Xv = (const float*)d_in[1];
  const float* Xq = (const float*)d_in[2];
  const float* Wk = (const float*)d_in[3];
  const float* Wv = (const float*)d_in[4];
  const float* Wq = (const float*)d_in[5];
  const float* Wo = (const float*)d_in[6];
  const float* bo = (const float*)d_in[7];
  float* out = (float*)d_out;

  // workspace partition (bf16): K, V, Q each [B*H][S][HD], concat [B][S][EMB]
  const size_t perTensor = (size_t)BATCH * HEADS * SEQ * HD;   // == B*S*EMB
  bf16_t* Kb = (bf16_t*)d_ws;
  bf16_t* Vb = Kb + perTensor;
  bf16_t* Qb = Vb + perTensor;
  bf16_t* Cc = Qb + perTensor;

  qkv_proj_kernel<<<dim3(SEQ / 16, HD / 16, 3 * BATCH * HEADS), 32, 0, stream>>>(
      Xk, Xv, Xq, Wk, Wv, Wq, Kb, Vb, Qb);
  flash_attn_kernel<<<dim3(SEQ / (16 * FA_WAVES), BATCH * HEADS), 32 * FA_WAVES,
                      0, stream>>>(Qb, Kb, Vb, Cc);
  out_proj_kernel<<<dim3(SEQ / 16, EMB / 16, BATCH), 32, 0, stream>>>(
      Cc, Wo, bo, out);
}
